// DebertaV2Attention_68899865362633
// MI455X (gfx1250) — matrix operand order repaired and checked by reading, not compiled, not run on hardware
//
#include <hip/hip_runtime.h>
#include <hip/hip_bf16.h>

// ---------------------------------------------------------------------------
// DeBERTa-v2 disentangled attention for MI455X (gfx1250, wave32, WMMA).
// Heavy math on v_wmma_f32_16x16x32_f16 (f16 in, f32 accumulate).
// GEMM: LDS-staged, double-buffered pipeline.  Tiles are staged by the gfx1250
// async DMA path (GLOBAL_LOAD_ASYNC_TO_LDS_B128, synchronized with
// s_wait_asynccnt) when the builtin is available; otherwise a register-staged
// copy is used.  Either way the next k-tile's loads overlap the current
// tile's 8 WMMAs per wave, and tiles are shared across all 8 waves of the
// block (~3x L2 traffic cut vs direct loads).  LDS rows padded to 40 halves
// so fragment reads are bank-conflict-free.  Workspace use: ~136 MB.
// ---------------------------------------------------------------------------

#define HID 1024
#define NSEQ 1024
#define NH 16
#define HD 64
#define RELROWS 2048

#if defined(__has_builtin)
#if __has_builtin(__builtin_amdgcn_global_load_async_to_lds_b128)
#define USE_ASYNC_LDS 1
#endif
#endif
#ifndef USE_ASYNC_LDS
#define USE_ASYNC_LDS 0
#endif

#define AS1 __attribute__((address_space(1)))
#define AS3 __attribute__((address_space(3)))

typedef _Float16 v16h __attribute__((ext_vector_type(16)));
typedef _Float16 v8h  __attribute__((ext_vector_type(8)));
typedef float    v8f  __attribute__((ext_vector_type(8)));
typedef int      v4i  __attribute__((vector_size(16)));   // matches builtin proto

// global (AS1) int4* from a flat pointer: flat == global address for AS1
__device__ __forceinline__ AS1 v4i* as_global_v4i(const void* p) {
    return (AS1 v4i*)(uintptr_t)p;
}
// LDS (AS3) int4* from a flat pointer: low 32 bits of a flat shared address
// are the LDS byte offset (ISA 10.2)
__device__ __forceinline__ AS3 v4i* as_lds_v4i(const void* p) {
    return (AS3 v4i*)(unsigned)(uintptr_t)p;
}

__device__ __forceinline__ int iclamp(int x, int lo, int hi) {
    return x < lo ? lo : (x > hi ? hi : x);
}

__device__ __forceinline__ void wait_async0() {
#if defined(__has_builtin) && __has_builtin(__builtin_amdgcn_s_wait_asynccnt)
    __builtin_amdgcn_s_wait_asynccnt(0);
#else
    asm volatile("s_wait_asynccnt 0x0" ::: "memory");
#endif
}

// A fragment: 16x32 f16 (M x K).  Contiguous k order, same as B (not the ISA table order):
//   lanes 0-15  : row M=lane,     halves K = {0..15}
//   lanes 16-31 : row M=lane-16,  halves K = {16..31}
__device__ __forceinline__ v16h load_a_frag(const _Float16* __restrict__ A,
                                            int row, int lda, int k0, int lane) {
    const _Float16* p = A + (size_t)row * lda + k0;
    int sel = (lane >> 4) & 1;
    union { v16h v; v8h h[2]; } u;
    u.h[0] = *(const v8h*)(p + sel * 16);
    u.h[1] = *(const v8h*)(p + sel * 16 + 8);
    return u.v;
}

// B fragment: 32x16 f16 (K x N), computing C = A * B^T with B stored (N,K)
// row-major. lane&15 selects column n (= row of B-matrix-as-stored), lanes
// 16-31 hold K=16..31.  16 contiguous halves per lane -> one 32B load.
__device__ __forceinline__ v16h load_b_frag(const _Float16* __restrict__ B,
                                            int row, int ldb, int k0, int lane) {
    const _Float16* p = B + (size_t)row * ldb + k0 + ((lane >> 4) & 1) * 16;
    return *(const v16h*)p;
}

__device__ __forceinline__ v8f wmma_f16(v16h a, v16h b, v8f c) {
    return __builtin_amdgcn_wmma_f32_16x16x32_f16(false, a, false, b,
                                                  (short)0, c, false, false);
}

// ---------------------------------------------------------------------------
// f32 -> f16 elementwise convert
// ---------------------------------------------------------------------------
__global__ void cvt_f32_f16_kernel(const float* __restrict__ in,
                                   _Float16* __restrict__ out, int n) {
    int i = blockIdx.x * blockDim.x + threadIdx.x;
    if (i < n) out[i] = (_Float16)in[i];
}

// ---------------------------------------------------------------------------
// Batched GEMM: C[z] = A[z] (M x Kd) @ B[z]^T (+ bias), B stored (N x Kd)
// row-major.  Block = WM*WN waves; block tile (WM*32) x (WN*64); per-wave
// tile 32x64 (8 f32 16x16 accumulators).  A and B k-tiles are staged through
// double-buffered LDS.
// ---------------------------------------------------------------------------
template <int WM, int WN>
__global__ __launch_bounds__(WM* WN * 32)
void gemm_bias_kernel(const _Float16* __restrict__ A, int lda, long long strideA,
                      const _Float16* __restrict__ B, int ldb, long long strideB,
                      const float* __restrict__ bias,
                      float* __restrict__ Cf, _Float16* __restrict__ Ch,
                      int ldc, long long strideC, int Kd) {
    constexpr int BM = WM * 32;           // block rows
    constexpr int BN = WN * 64;           // block cols
    constexpr int NT = WM * WN * 32;      // threads per block
    constexpr int AU = (BM * 4) / NT;     // 16B A chunks per thread
    constexpr int BU = (BN * 4) / NT;     // 16B B chunks per thread
    constexpr int ROWH = 40;              // padded LDS row stride (halves)

    __shared__ _Float16 smem[2][(BM + BN) * ROWH];

    const int tid  = threadIdx.x;
    const int lane = tid & 31;
    const int wave = tid >> 5;
    const int wm = wave % WM;
    const int wn = wave / WM;
    const int z = blockIdx.z;
    const _Float16* Az = A + (size_t)z * strideA + (size_t)(blockIdx.y * BM) * lda;
    const _Float16* Bz = B + (size_t)z * strideB + (size_t)(blockIdx.x * BN) * ldb;

    auto lds_a_frag = [&](int buf, int row) -> v16h {
        const _Float16* p = &smem[buf][row * ROWH];
        const int sel = (lane >> 4) & 1;
        union { v16h v; v8h h[2]; } u;
        u.h[0] = *(const v8h*)(p + sel * 16);
        u.h[1] = *(const v8h*)(p + sel * 16 + 8);
        return u.v;
    };
    auto lds_b_frag = [&](int buf, int row) -> v16h {
        const _Float16* p = &smem[buf][(BM + row) * ROWH + ((lane >> 4) & 1) * 16];
        return *(const v16h*)p;
    };

    const int m0 = wm * 32;               // within block tile
    const int n0 = wn * 64;

    v8f acc[2][4];
#pragma unroll
    for (int mi = 0; mi < 2; ++mi)
#pragma unroll
        for (int t = 0; t < 4; ++t) acc[mi][t] = v8f{};

    auto compute = [&](int cur) {
        v16h a0 = lds_a_frag(cur, m0 + (lane & 15));
        v16h a1 = lds_a_frag(cur, m0 + 16 + (lane & 15));
        v16h bf[4];
#pragma unroll
        for (int t = 0; t < 4; ++t)
            bf[t] = lds_b_frag(cur, n0 + t * 16 + (lane & 15));
#pragma unroll
        for (int t = 0; t < 4; ++t) {
            acc[0][t] = wmma_f16(a0, bf[t], acc[0][t]);
            acc[1][t] = wmma_f16(a1, bf[t], acc[1][t]);
        }
    };

    const int nk = Kd >> 5;

#if USE_ASYNC_LDS
    // --------- async DMA staging: global -> LDS without touching VGPRs ------
    auto stage = [&](int buf, int k0) {
#pragma unroll
        for (int i = 0; i < AU; ++i) {
            const int u = tid + NT * i, row = u >> 2, ch = u & 3;
            __builtin_amdgcn_global_load_async_to_lds_b128(
                as_global_v4i(Az + (size_t)row * lda + k0 + ch * 8),
                as_lds_v4i(&smem[buf][row * ROWH + ch * 8]), 0, 0);
        }
#pragma unroll
        for (int i = 0; i < BU; ++i) {
            const int u = tid + NT * i, row = u >> 2, ch = u & 3;
            __builtin_amdgcn_global_load_async_to_lds_b128(
                as_global_v4i(Bz + (size_t)row * ldb + k0 + ch * 8),
                as_lds_v4i(&smem[buf][(BM + row) * ROWH + ch * 8]), 0, 0);
        }
    };

    stage(0, 0);
    wait_async0();
    __syncthreads();
    for (int ki = 0; ki < nk; ++ki) {
        const int cur = ki & 1;
        if (ki + 1 < nk) stage(1 - cur, (ki + 1) << 5);  // DMA in flight
        compute(cur);
        if (ki + 1 < nk) {
            wait_async0();                // this wave's DMA writes landed
            __syncthreads();              // all waves' writes + all reads done
        }
    }
#else
    // --------- fallback: register-staged copy (global->reg->LDS) -----------
    v8h aR[AU], bR[BU];
    auto ldreg = [&](int k0) {
#pragma unroll
        for (int i = 0; i < AU; ++i) {
            const int u = tid + NT * i, row = u >> 2, ch = u & 3;
            aR[i] = *(const v8h*)(Az + (size_t)row * lda + k0 + ch * 8);
        }
#pragma unroll
        for (int i = 0; i < BU; ++i) {
            const int u = tid + NT * i, row = u >> 2, ch = u & 3;
            bR[i] = *(const v8h*)(Bz + (size_t)row * ldb + k0 + ch * 8);
        }
    };
    auto streg = [&](int buf) {
#pragma unroll
        for (int i = 0; i < AU; ++i) {
            const int u = tid + NT * i, row = u >> 2, ch = u & 3;
            *(v8h*)(&smem[buf][row * ROWH + ch * 8]) = aR[i];
        }
#pragma unroll
        for (int i = 0; i < BU; ++i) {
            const int u = tid + NT * i, row = u >> 2, ch = u & 3;
            *(v8h*)(&smem[buf][(BM + row) * ROWH + ch * 8]) = bR[i];
        }
    };

    ldreg(0);
    streg(0);
    __syncthreads();
    for (int ki = 0; ki < nk; ++ki) {
        const int cur = ki & 1;
        if (ki + 1 < nk) ldreg((ki + 1) << 5);   // next k-tile in flight
        compute(cur);
        if (ki + 1 < nk) {
            __syncthreads();                      // everyone done reading buf[1-cur]
            streg(1 - cur);                       // commit next tile
            __syncthreads();                      // next tile visible to all
        }
    }
#endif

    const int mrow = ((lane >> 4) & 1) * 8;
#pragma unroll
    for (int mi = 0; mi < 2; ++mi) {
#pragma unroll
        for (int t = 0; t < 4; ++t) {
            const int n = blockIdx.x * BN + n0 + t * 16 + (lane & 15);
            const float bv = bias ? bias[n] : 0.0f;
#pragma unroll
            for (int vv = 0; vv < 8; ++vv) {
                const int m = blockIdx.y * BM + m0 + mi * 16 + mrow + vv;
                const float val = acc[mi][t][vv] + bv;
                const size_t off = (size_t)z * strideC + (size_t)m * ldc + n;
                if (Cf) Cf[off] = val;
                if (Ch) Ch[off] = (_Float16)val;
            }
        }
    }
}

// ---------------------------------------------------------------------------
// f16 1024x1024 transpose (for V^T so attn@V becomes the standard A@B^T form)
// ---------------------------------------------------------------------------
__global__ void transpose_f16_kernel(const _Float16* __restrict__ in,
                                     _Float16* __restrict__ out) {
    const int r = blockIdx.x;
    for (int c = threadIdx.x; c < HID; c += blockDim.x)
        out[(size_t)c * HID + r] = in[(size_t)r * HID + c];
}

// ---------------------------------------------------------------------------
// Fused score kernel.  scores[h,i,j] =
//   ( q_h[i].k_h[j] + q_h[i].pos_k_h[1024+i-j] + k_h[j].pos_q_h[1024+i-j] ) / sqrt(192)
// Each wave owns one 16x16 (i,j) tile.  The two relative terms are computed
// as 16x32 WMMA GEMMs against the 31-row pos window (base-15..base+15) and
// then gathered along the tile diagonals through LDS.
// ---------------------------------------------------------------------------
__global__ void scores_kernel(const _Float16* __restrict__ Q,
                              const _Float16* __restrict__ Kk,
                              const _Float16* __restrict__ PK,
                              const _Float16* __restrict__ PQ,
                              float* __restrict__ scores, float rscale) {
    const int lane = threadIdx.x & 31;
    const int wave = threadIdx.x >> 5;
    const int h  = blockIdx.z;
    const int i0 = blockIdx.y * 16;
    const int j0 = blockIdx.x * 128 + wave * 16;

    const _Float16* q  = Q  + h * HD;   // row stride HID
    const _Float16* kk = Kk + h * HD;
    const _Float16* pk = PK + h * HD;
    const _Float16* pq = PQ + h * HD;

    const int base = 1024 + i0 - j0;    // pos row at (a-b)==0

    v8f sacc = v8f{};
    v8f m1[2] = {v8f{}, v8f{}};
    v8f m2[2] = {v8f{}, v8f{}};

#pragma unroll
    for (int k0 = 0; k0 < HD; k0 += 32) {
        v16h aq = load_a_frag(q,  i0 + (lane & 15), HID, k0, lane);
        v16h ak = load_a_frag(kk, j0 + (lane & 15), HID, k0, lane);
        v16h bk = load_b_frag(kk, j0 + (lane & 15), HID, k0, lane);
        sacc = wmma_f16(aq, bk, sacc);
#pragma unroll
        for (int t = 0; t < 2; ++t) {
            const int prow = iclamp(base - 15 + t * 16 + (lane & 15), 0, RELROWS - 1);
            v16h bpk = load_b_frag(pk, prow, HID, k0, lane);
            m1[t] = wmma_f16(aq, bpk, m1[t]);
            v16h bpq = load_b_frag(pq, prow, HID, k0, lane);
            m2[t] = wmma_f16(ak, bpq, m2[t]);
        }
    }

    // per-wave LDS scratch: M1 (16x32) and M2 (16x32) f32
    __shared__ float lds[8][2][16][32];  // 32 KB
    const int mrow = ((lane >> 4) & 1) * 8;
#pragma unroll
    for (int t = 0; t < 2; ++t) {
        const int col = t * 16 + (lane & 15);
#pragma unroll
        for (int vv = 0; vv < 8; ++vv) {
            lds[wave][0][mrow + vv][col] = m1[t][vv];
            lds[wave][1][mrow + vv][col] = m2[t][vv];
        }
    }
    __syncthreads();

    const int n = lane & 15;
#pragma unroll
    for (int vv = 0; vv < 8; ++vv) {
        const int m = mrow + vv;
        const int t = m - n + 15;                       // 0..30
        const float c2p = lds[wave][0][m][t];
        const float p2c = lds[wave][1][n][t];
        const float val = (sacc[vv] + c2p + p2c) * rscale;
        scores[((size_t)(h * NSEQ + i0 + m)) * NSEQ + (j0 + n)] = val;
    }
}

// ---------------------------------------------------------------------------
// Row softmax (exact two-pass, f32 in, f16 out). One block per (h,i) row.
// ---------------------------------------------------------------------------
__global__ void softmax_kernel(const float* __restrict__ scores,
                               _Float16* __restrict__ attn) {
    const int row = blockIdx.x;
    const float* s = scores + (size_t)row * NSEQ;
    _Float16* a = attn + (size_t)row * NSEQ;
    __shared__ float red[256];
    const int tid = threadIdx.x;

    float mx = -3.0e38f;
    for (int j = tid; j < NSEQ; j += 256) mx = fmaxf(mx, s[j]);
    red[tid] = mx; __syncthreads();
    for (int st = 128; st > 0; st >>= 1) {
        if (tid < st) red[tid] = fmaxf(red[tid], red[tid + st]);
        __syncthreads();
    }
    mx = red[0]; __syncthreads();

    float sum = 0.0f;
    for (int j = tid; j < NSEQ; j += 256) sum += __expf(s[j] - mx);
    red[tid] = sum; __syncthreads();
    for (int st = 128; st > 0; st >>= 1) {
        if (tid < st) red[tid] += red[tid + st];
        __syncthreads();
    }
    const float inv = 1.0f / red[0];
    for (int j = tid; j < NSEQ; j += 256)
        a[j] = (_Float16)(__expf(s[j] - mx) * inv);
}

// ---------------------------------------------------------------------------
// Residual + LayerNorm: y = LN(out + hidden) * g + b.  One block per row.
// ---------------------------------------------------------------------------
__global__ void resid_ln_kernel(const float* __restrict__ o,
                                const float* __restrict__ hs,
                                const float* __restrict__ g,
                                const float* __restrict__ b,
                                float* __restrict__ y) {
    const int row = blockIdx.x;
    const int tid = threadIdx.x;
    __shared__ float xs[HID];
    __shared__ float red[256];

    float lsum = 0.0f;
    for (int c = tid; c < HID; c += 256) {
        const float x = o[(size_t)row * HID + c] + hs[(size_t)row * HID + c];
        xs[c] = x;
        lsum += x;
    }
    red[tid] = lsum; __syncthreads();
    for (int st = 128; st > 0; st >>= 1) {
        if (tid < st) red[tid] += red[tid + st];
        __syncthreads();
    }
    const float mu = red[0] * (1.0f / HID);
    __syncthreads();

    float lvar = 0.0f;
    for (int c = tid; c < HID; c += 256) {
        const float d = xs[c] - mu;
        lvar += d * d;
    }
    red[tid] = lvar; __syncthreads();
    for (int st = 128; st > 0; st >>= 1) {
        if (tid < st) red[tid] += red[tid + st];
        __syncthreads();
    }
    const float rstd = rsqrtf(red[0] * (1.0f / HID) + 1e-7f);
    __syncthreads();
    for (int c = tid; c < HID; c += 256)
        y[(size_t)row * HID + c] = (xs[c] - mu) * rstd * g[c] + b[c];
}

// ---------------------------------------------------------------------------
// Host-side orchestration
// ---------------------------------------------------------------------------
static inline void cvt(const float* in, _Float16* out, int n, hipStream_t s) {
    cvt_f32_f16_kernel<<<(n + 255) / 256, 256, 0, s>>>(in, out, n);
}

extern "C" void kernel_launch(void* const* d_in, const int* in_sizes, int n_in,
                              void* d_out, int out_size, void* d_ws, size_t ws_size,
                              hipStream_t stream) {
    const float* hs   = (const float*)d_in[0];
    const float* rel  = (const float*)d_in[1];
    const float* Wq   = (const float*)d_in[2];
    const float* bq   = (const float*)d_in[3];
    const float* Wk   = (const float*)d_in[4];
    const float* bk   = (const float*)d_in[5];
    const float* Wv   = (const float*)d_in[6];
    const float* bv   = (const float*)d_in[7];
    const float* Wpk  = (const float*)d_in[8];
    const float* bpk  = (const float*)d_in[9];
    const float* Wpq  = (const float*)d_in[10];
    const float* bpq  = (const float*)d_in[11];
    const float* Wo   = (const float*)d_in[12];
    const float* bo   = (const float*)d_in[13];
    const float* ln_g = (const float*)d_in[14];
    const float* ln_b = (const float*)d_in[15];
    float* out = (float*)d_out;

    // workspace carve-up (MB-aligned segments, total ~136 MB)
    char* ws = (char*)d_ws;
    const size_t MB = 1024ull * 1024ull;
    _Float16* hsf   = (_Float16*)(ws + 0 * MB);    // 2 MB
    _Float16* relf  = (_Float16*)(ws + 2 * MB);    // 4 MB
    _Float16* wq    = (_Float16*)(ws + 6 * MB);    // 2 MB each
    _Float16* wk    = (_Float16*)(ws + 8 * MB);
    _Float16* wv    = (_Float16*)(ws + 10 * MB);
    _Float16* wpk   = (_Float16*)(ws + 12 * MB);
    _Float16* wpq   = (_Float16*)(ws + 14 * MB);
    _Float16* wo    = (_Float16*)(ws + 16 * MB);
    _Float16* qf    = (_Float16*)(ws + 18 * MB);
    _Float16* kf    = (_Float16*)(ws + 20 * MB);
    _Float16* vf    = (_Float16*)(ws + 22 * MB);
    _Float16* vt    = (_Float16*)(ws + 24 * MB);
    _Float16* posk  = (_Float16*)(ws + 26 * MB);   // 4 MB
    _Float16* posq  = (_Float16*)(ws + 30 * MB);   // 4 MB
    _Float16* ctx   = (_Float16*)(ws + 34 * MB);   // 2 MB
    float*    outp  = (float*)   (ws + 36 * MB);   // 4 MB
    float*    sc    = (float*)   (ws + 40 * MB);   // 64 MB
    _Float16* attn  = (_Float16*)(ws + 104 * MB);  // 32 MB

    // 1) convert activations + weights to f16
    cvt(hs,  hsf,  NSEQ * HID, stream);
    cvt(rel, relf, RELROWS * HID, stream);
    cvt(Wq, wq, HID * HID, stream);
    cvt(Wk, wk, HID * HID, stream);
    cvt(Wv, wv, HID * HID, stream);
    cvt(Wpk, wpk, HID * HID, stream);
    cvt(Wpq, wpq, HID * HID, stream);
    cvt(Wo, wo, HID * HID, stream);

    dim3 blk(256);

    // 2) projections: X @ W^T + b  (M=1024 or 2048, N=K=1024)
    {
        dim3 g(HID / 128, NSEQ / 128, 1);
        gemm_bias_kernel<4, 2><<<g, blk, 0, stream>>>(hsf, HID, 0, wq, HID, 0, bq,
                                                      nullptr, qf, HID, 0, HID);
        gemm_bias_kernel<4, 2><<<g, blk, 0, stream>>>(hsf, HID, 0, wk, HID, 0, bk,
                                                      nullptr, kf, HID, 0, HID);
        gemm_bias_kernel<4, 2><<<g, blk, 0, stream>>>(hsf, HID, 0, wv, HID, 0, bv,
                                                      nullptr, vf, HID, 0, HID);
        dim3 g2(HID / 128, RELROWS / 128, 1);
        gemm_bias_kernel<4, 2><<<g2, blk, 0, stream>>>(relf, HID, 0, wpk, HID, 0, bpk,
                                                       nullptr, posk, HID, 0, HID);
        gemm_bias_kernel<4, 2><<<g2, blk, 0, stream>>>(relf, HID, 0, wpq, HID, 0, bpq,
                                                       nullptr, posq, HID, 0, HID);
    }

    // 3) V^T so that attn@V maps onto the A@B^T GEMM
    transpose_f16_kernel<<<HID, blk, 0, stream>>>(vf, vt);

    // 4) fused scores: q.k^T + c2p + p2c, all via WMMA
    {
        const float rscale = 1.0f / sqrtf((float)(HD * 3));
        dim3 g(NSEQ / 128, NSEQ / 16, NH);
        scores_kernel<<<g, blk, 0, stream>>>(qf, kf, posk, posq, sc, rscale);
    }

    // 5) softmax over each of the 16*1024 rows
    softmax_kernel<<<NH * NSEQ, blk, 0, stream>>>(sc, attn);

    // 6) ctx_h = attn_h @ V_h : batched over heads (N=64 per head)
    {
        dim3 g(HD / 64, NSEQ / 256, NH);
        gemm_bias_kernel<8, 1><<<g, blk, 0, stream>>>(
            attn, NSEQ, (long long)NSEQ * NSEQ,       // A: per-head 1024x1024
            vt, NSEQ, (long long)HD * NSEQ,           // B: per-head 64x1024 (V^T)
            nullptr,
            nullptr, ctx, HID, (long long)HD,         // C: cols h*64..h*64+63
            NSEQ);
    }

    // 7) output projection: ctx @ Wo^T + bo (f32 out)
    {
        dim3 g(HID / 128, NSEQ / 128, 1);
        gemm_bias_kernel<4, 2><<<g, blk, 0, stream>>>(ctx, HID, 0, wo, HID, 0, bo,
                                                      outp, nullptr, HID, 0, HID);
    }

    // 8) residual + LayerNorm -> d_out
    resid_ln_kernel<<<NSEQ, blk, 0, stream>>>(outp, hs, ln_g, ln_b, out);

    (void)in_sizes; (void)n_in; (void)out_size; (void)ws_size;
}
